// SSD_12154757447772
// MI455X (gfx1250) — compile-verified, matrix operand order checked by reading, not run on hardware
//
#include <hip/hip_runtime.h>
#include <hip/hip_bf16.h>

// ---------------------------------------------------------------------------
// Types for CDNA5 WMMA (gfx1250, wave32)
// ---------------------------------------------------------------------------
typedef __attribute__((ext_vector_type(16))) __bf16 v16bf;
typedef __attribute__((ext_vector_type(8)))  float  v8f;

__device__ __forceinline__ __bf16 f2bf(float f) {
  unsigned int u = __builtin_bit_cast(unsigned int, f);
  unsigned int r = u + 0x7FFFu + ((u >> 16) & 1u);   // round-to-nearest-even
  unsigned short h = (unsigned short)(r >> 16);
  return __builtin_bit_cast(__bf16, h);
}
__device__ __forceinline__ __bf16 bf_zero() {
  return __builtin_bit_cast(__bf16, (unsigned short)0);
}

#define NEG_BIG -1.0e30f
#define LDS_STR 34   // padded K-stride for staged B panel (bank-conflict avoid)

// ---------------------------------------------------------------------------
// Weight pack: fp32 OIHW -> bf16 [K][Cout]  (K = Cin*k*k)
// ---------------------------------------------------------------------------
__global__ void pack_weights(const float* __restrict__ w, __bf16* __restrict__ Bp,
                             int K, int Cout) {
  int i = blockIdx.x * blockDim.x + threadIdx.x;
  if (i >= K * Cout) return;
  int k = i / Cout, o = i - k * Cout;
  Bp[i] = f2bf(w[(size_t)o * K + k]);
}

// ---------------------------------------------------------------------------
// Implicit-GEMM convolution with bf16 WMMA (fp32 accumulate).
// Block = 8 waves sharing one 128(M) x 64(N) output tile of image n.
// Per 32-K chunk the block stages the 32x64 bf16 weight panel into LDS
// (N-major, padded stride) once; each wave then assembles B fragments from
// LDS and its private A fragment by global gather, issuing 4
// v_wmma_f32_16x16x32_bf16 per chunk (A reused across the 4 N sub-tiles).
// Fragment layouts per CDNA5 ISA 7.12.2.  KSZ templated so the kk->(ci,ky,kx)
// decomposition uses constant-divisor strength reduction.
// ---------------------------------------------------------------------------
template <int KSZ>
__global__ __launch_bounds__(256) void conv_wmma(
    const float* __restrict__ in, const __bf16* __restrict__ Bp,
    const float* __restrict__ bias, float* __restrict__ out,
    int N, int Cin, int Hin, int Win, int Cout, int Hout, int Wout,
    int stride, int pad, int dil, int relu) {
  constexpr int KK2 = KSZ * KSZ;
  const int M   = Hout * Wout;
  const int K   = Cin * KK2;
  const int tMg = (M + 127) >> 7;        // 128 M-rows per block
  const int tN  = (Cout + 63) >> 6;

  int bid = blockIdx.x;
  int nt  = bid % tN;  int t2 = bid / tN;
  int mg  = t2 % tMg;  int n  = t2 / tMg;

  const int wv   = threadIdx.x >> 5;     // wave in block: M sub-tile
  const int lane = threadIdx.x & 31;
  const int hh   = lane >> 4;            // lane half (0/1)
  const int l    = lane & 15;
  const int cb   = nt * 64;
  const int mt   = mg * 8 + wv;

  const int  arow = mt * 16 + l;         // A row this lane loads
  const bool am   = arow < M;
  const int  oh   = am ? arow / Wout : 0;
  const int  ow   = am ? arow - oh * Wout : 0;
  const float* inN = in + (size_t)n * Cin * Hin * Win;

  // K-local index of fragment element e (ISA 7.12.2, 16-bit A 16x32):
  // lanes 0-15 hold K={0..7,16..23}; lanes 16-31 hold K={8..15,24..31}
  int klocal[16];
#pragma unroll
  for (int e = 0; e < 16; e++) {
    int v = e >> 1, p = e & 1;
    klocal[e] = ((v >> 2) << 4) + hh * 8 + ((v & 3) << 1) + p;
  }

  __shared__ __bf16 ldsB[64 * LDS_STR]; // [col][k] padded
  const __bf16 bz = bf_zero();
  v8f c0 = {}, c1 = {}, c2 = {}, c3 = {};

  for (int kb = 0; kb < K; kb += 32) {
    // ---- cooperative stage of B chunk into LDS (zero-padded) ----
    __syncthreads();
#pragma unroll
    for (int q = 0; q < 8; q++) {
      int li  = threadIdx.x * 8 + q;     // 0..2047
      int k   = li >> 6;                 // 0..31
      int col = li & 63;
      int kk  = kb + k;
      int c   = cb + col;
      __bf16 v = (kk < K && c < Cout) ? Bp[(size_t)kk * Cout + c] : bz;
      ldsB[col * LDS_STR + k] = v;
    }
    __syncthreads();
    if (kb + 32 < K)                     // prefetch next global B panel
      __builtin_prefetch(Bp + (size_t)(kb + 32) * Cout + cb + l, 0, 1);

    // ---- A fragment: per-lane global gather with bf16 convert ----
    v16bf a, b0, b1, b2, b3;
#pragma unroll
    for (int e = 0; e < 16; e++) {
      const int kk = kb + klocal[e];
      __bf16 av = bz;
      if (am && kk < K) {
        int ci = kk / KK2;               // constant-divisor
        int r2 = kk - ci * KK2;
        int ky = (KSZ == 1) ? 0 : r2 / KSZ;
        int kx = (KSZ == 1) ? 0 : r2 - ky * KSZ;
        int ih = oh * stride - pad + ky * dil;
        int iw = ow * stride - pad + kx * dil;
        if ((unsigned)ih < (unsigned)Hin && (unsigned)iw < (unsigned)Win)
          av = f2bf(inN[((size_t)ci * Hin + ih) * Win + iw]);
      }
      a[e] = av;
      // ---- B fragments straight from LDS (already zero-padded) ----
      const int kl = klocal[e];
      b0[e] = ldsB[(l     ) * LDS_STR + kl];
      b1[e] = ldsB[(l + 16) * LDS_STR + kl];
      b2[e] = ldsB[(l + 32) * LDS_STR + kl];
      b3[e] = ldsB[(l + 48) * LDS_STR + kl];
    }
    c0 = __builtin_amdgcn_wmma_f32_16x16x32_bf16(false, a, false, b0, (short)0, c0, false, false);
    c1 = __builtin_amdgcn_wmma_f32_16x16x32_bf16(false, a, false, b1, (short)0, c1, false, false);
    c2 = __builtin_amdgcn_wmma_f32_16x16x32_bf16(false, a, false, b2, (short)0, c2, false, false);
    c3 = __builtin_amdgcn_wmma_f32_16x16x32_bf16(false, a, false, b3, (short)0, c3, false, false);
  }

  // Epilogue: C/D layout — VGPR r: lanes0-15 -> M=r, lanes16-31 -> M=r+8
#pragma unroll
  for (int r = 0; r < 8; r++) {
    int m = mt * 16 + r + hh * 8;
    if (m >= M) continue;
    int mh = m / Wout, mw = m - mh * Wout;
    float vals[4] = {c0[r], c1[r], c2[r], c3[r]};
#pragma unroll
    for (int j = 0; j < 4; j++) {
      int col = cb + j * 16 + l;
      if (col < Cout) {
        float v = vals[j] + bias[col];
        if (relu) v = v > 0.f ? v : 0.f;
        out[(((size_t)n * Cout + col) * Hout + mh) * Wout + mw] = v;
      }
    }
  }
}

// ---------------------------------------------------------------------------
// Max pool (supports asymmetric ceil-mode padding via padLo + explicit Hout)
// ---------------------------------------------------------------------------
__global__ void maxpool_k(const float* __restrict__ in, float* __restrict__ out,
                          int N, int C, int Hin, int Win, int Hout, int Wout,
                          int k, int s, int plo) {
  int i = blockIdx.x * blockDim.x + threadIdx.x;
  int tot = N * C * Hout * Wout;
  if (i >= tot) return;
  int ww = i % Wout, t = i / Wout;
  int hq = t % Hout; t /= Hout;
  int c = t % C; int n = t / C;
  const float* base = in + ((size_t)n * C + c) * Hin * Win;
  float m = NEG_BIG;
  for (int ky = 0; ky < k; ky++) {
    int ih = hq * s - plo + ky;
    if ((unsigned)ih >= (unsigned)Hin) continue;
    for (int kx = 0; kx < k; kx++) {
      int iw = ww * s - plo + kx;
      if ((unsigned)iw >= (unsigned)Win) continue;
      float v = base[(size_t)ih * Win + iw];
      m = v > m ? v : m;
    }
  }
  out[i] = m;
}

// ---------------------------------------------------------------------------
// L2 normalize over channels + learned per-channel scale (conv4_3)
// ---------------------------------------------------------------------------
__global__ void l2norm_k(const float* __restrict__ in, float* __restrict__ out,
                         const float* __restrict__ scale, int N, int C, int HW) {
  int i = blockIdx.x * blockDim.x + threadIdx.x;
  if (i >= N * HW) return;
  int n = i / HW, p = i - n * HW;
  const float* b = in + ((size_t)n * C) * HW + p;
  float ss = 0.f;
  for (int c = 0; c < C; c++) { float v = b[(size_t)c * HW]; ss += v * v; }
  float nor = sqrtf(ss);
  nor = nor > 1e-12f ? nor : 1e-12f;
  float inv = 1.0f / nor;
  float* o = out + ((size_t)n * C) * HW + p;
  for (int c = 0; c < C; c++) o[(size_t)c * HW] = b[(size_t)c * HW] * inv * scale[c];
}

// ---------------------------------------------------------------------------
// Head scatter: NCHW (C = A*T) -> [N, PTOT, T] at prior offset
// prior = off + (h*W+w)*A + a ; task index t
// ---------------------------------------------------------------------------
__global__ void scatter_head(const float* __restrict__ src, float* __restrict__ dst,
                             int N, int A, int T, int H, int W, int off, int PTOT) {
  int i = blockIdx.x * blockDim.x + threadIdx.x;
  int C = A * T;
  int tot = N * C * H * W;
  if (i >= tot) return;
  int w = i % W, t2 = i / W;
  int hq = t2 % H; t2 /= H;
  int c = t2 % C; int n = t2 / C;
  int a = c / T, t = c - a * T;
  size_t di = ((size_t)n * PTOT + off + ((size_t)hq * W + w) * A + a) * T + t;
  dst[di] = src[i];
}

// ---------------------------------------------------------------------------
// Softmax over 81 classes, keep first 80 (background last)
// ---------------------------------------------------------------------------
__global__ void softmax_k(const float* __restrict__ cls, float* __restrict__ scr,
                          int NP) {
  int i = blockIdx.x * blockDim.x + threadIdx.x;
  if (i >= NP) return;
  const float* r = cls + (size_t)i * 81;
  float mx = r[0];
  for (int c = 1; c < 81; c++) mx = r[c] > mx ? r[c] : mx;
  float s = 0.f;
  for (int c = 0; c < 81; c++) s += expf(r[c] - mx);
  float inv = 1.0f / s;
  float* o = scr + (size_t)i * 80;
  for (int c = 0; c < 80; c++) o[c] = expf(r[c] - mx) * inv;
}

// ---------------------------------------------------------------------------
// Prior (anchor) generation, replicating _build_priors. priors[p] = cx,cy,w,h
// ---------------------------------------------------------------------------
__global__ void priors_k(float* __restrict__ pri) {
  const int   fs[6]  = {38, 19, 10, 5, 3, 1};
  const int   st[6]  = {8, 16, 32, 64, 100, 300};
  const int   an[6]  = {4, 6, 6, 6, 4, 4};
  const int   off[6] = {0, 5776, 7942, 8542, 8692, 8728};
  const float sz[7]  = {21.f, 45.f, 99.f, 153.f, 207.f, 261.f, 315.f};
  int p = blockIdx.x * blockDim.x + threadIdx.x;
  if (p >= 8732) return;
  int L = 5;
  for (int i = 0; i < 5; i++) { if (p < off[i + 1]) { L = i; break; } }
  int li = p - off[L];
  int A = an[L];
  int cell = li / A, a = li - cell * A;
  int f = fs[L];
  int y = cell / f, x = cell - y * f;
  float cx = (x + 0.5f) * st[L];
  float cy = (y + 0.5f) * st[L];
  float s = sz[L];
  const float SQ2 = 1.41421356237f, SQ3 = 1.73205080757f;
  float w, h2;
  if (a == 0)      { w = s;        h2 = s; }
  else if (a == 1) { float e = sqrtf(s * sz[L + 1]); w = e; h2 = e; }
  else if (a == 2) { w = s * SQ2;  h2 = s / SQ2; }
  else if (a == 3) { w = s / SQ2;  h2 = s * SQ2; }
  else if (a == 4) { w = s * SQ3;  h2 = s / SQ3; }
  else             { w = s / SQ3;  h2 = s * SQ3; }
  pri[p * 4 + 0] = cx; pri[p * 4 + 1] = cy;
  pri[p * 4 + 2] = w;  pri[p * 4 + 3] = h2;
}

// ---------------------------------------------------------------------------
// Threshold + decode -> per-image candidate list {x1,y1,x2,y2,score,cls}
// ---------------------------------------------------------------------------
#define CAND_CAP 16384
__global__ void decode_k(const float* __restrict__ scr, const float* __restrict__ reg,
                         const float* __restrict__ pri, float* __restrict__ cand,
                         int* __restrict__ cnt, int N, int P) {
  int i = blockIdx.x * blockDim.x + threadIdx.x;
  long tot = (long)N * P * 80;
  if (i >= tot) return;
  int cls = i % 80;
  int t = i / 80;
  int p = t % P, n = t / P;
  float sc = scr[(size_t)(n * P + p) * 80 + cls];
  if (!(sc > 0.02f)) return;
  const float* pr = pri + (size_t)p * 4;
  const float* rg = reg + (size_t)(n * P + p) * 4;
  const float LOGWH = 4.13516655674f; // log(1000/16)
  float cx = pr[0] + rg[0] * 0.1f * pr[2];
  float cy = pr[1] + rg[1] * 0.1f * pr[3];
  float ew = rg[2] * 0.2f; ew = ew < LOGWH ? ew : LOGWH;
  float eh = rg[3] * 0.2f; eh = eh < LOGWH ? eh : LOGWH;
  float w = pr[2] * expf(ew);
  float h = pr[3] * expf(eh);
  float x1 = cx - w * 0.5f, y1 = cy - h * 0.5f;
  float x2 = cx + w * 0.5f, y2 = cy + h * 0.5f;
  x1 = fminf(fmaxf(x1, 0.f), 300.f); y1 = fminf(fmaxf(y1, 0.f), 300.f);
  x2 = fminf(fmaxf(x2, 0.f), 300.f); y2 = fminf(fmaxf(y2, 0.f), 300.f);
  if (!(x2 > x1 && y2 > y1)) return;
  int idx = atomicAdd(&cnt[n], 1);
  if (idx >= CAND_CAP) return;
  float* c = cand + ((size_t)n * CAND_CAP + idx) * 6;
  c[0] = x1; c[1] = y1; c[2] = x2; c[3] = y2; c[4] = sc; c[5] = (float)cls;
}

// ---------------------------------------------------------------------------
// Greedy per-image NMS: 200 picks, class-aware (equivalent to offset trick),
// block-wide argmax reduction in LDS.
// ---------------------------------------------------------------------------
__global__ __launch_bounds__(256) void nms_k(float* __restrict__ cand,
                                             const int* __restrict__ cnt,
                                             float* __restrict__ oboxes,
                                             float* __restrict__ oscores,
                                             float* __restrict__ oclasses) {
  __shared__ float sv[256];
  __shared__ int   si[256];
  int n   = blockIdx.x;
  int tid = threadIdx.x;
  int m   = cnt[n]; if (m > CAND_CAP) m = CAND_CAP;
  float* C = cand + (size_t)n * CAND_CAP * 6;

  for (int it = 0; it < 200; it++) {
    float best = NEG_BIG; int bi = -1;
    for (int j = tid; j < m; j += 256) {
      float s = C[(size_t)j * 6 + 4];
      if (s > best) { best = s; bi = j; }
    }
    sv[tid] = best; si[tid] = bi;
    __syncthreads();
    for (int o = 128; o > 0; o >>= 1) {
      if (tid < o && sv[tid + o] > sv[tid]) { sv[tid] = sv[tid + o]; si[tid] = si[tid + o]; }
      __syncthreads();
    }
    float bs = sv[0]; int bidx = si[0];
    __syncthreads();
    if (!(bs > 0.02f) || bidx < 0) break;   // uniform across block

    float bx1 = C[(size_t)bidx * 6 + 0], by1 = C[(size_t)bidx * 6 + 1];
    float bx2 = C[(size_t)bidx * 6 + 2], by2 = C[(size_t)bidx * 6 + 3];
    float bcl = C[(size_t)bidx * 6 + 5];
    if (tid == 0) {
      size_t ob = ((size_t)n * 200 + it) * 4;
      oboxes[ob + 0] = bx1; oboxes[ob + 1] = by1;
      oboxes[ob + 2] = bx2; oboxes[ob + 3] = by2;
      oscores[(size_t)n * 200 + it]  = bs;
      oclasses[(size_t)n * 200 + it] = bcl;
    }
    float barea = (bx2 - bx1) * (by2 - by1);
    for (int j = tid; j < m; j += 256) {
      float* cj = C + (size_t)j * 6;
      if ((int)cj[5] != (int)bcl) continue;
      float xx1 = fmaxf(bx1, cj[0]), yy1 = fmaxf(by1, cj[1]);
      float xx2 = fminf(bx2, cj[2]), yy2 = fminf(by2, cj[3]);
      float inter = fmaxf(xx2 - xx1, 0.f) * fmaxf(yy2 - yy1, 0.f);
      float aj = (cj[2] - cj[0]) * (cj[3] - cj[1]);
      float iou = inter / (barea + aj - inter + 1e-9f);
      if (iou > 0.45f) cj[4] = NEG_BIG;
    }
    __syncthreads();
  }
}

// ---------------------------------------------------------------------------
// Host orchestration
// ---------------------------------------------------------------------------
extern "C" void kernel_launch(void* const* d_in, const int* in_sizes, int n_in,
                              void* d_out, int out_size, void* d_ws, size_t ws_size,
                              hipStream_t stream) {
  (void)in_sizes; (void)ws_size;
  if (n_in < 72) return;
  const int B = 8;

  const float* x = (const float*)d_in[0];
  const float *Wv[15], *Bv[15], *We[8], *Be[8], *Wc[6], *Bc[6], *Wr[6], *Br[6];
  for (int i = 0; i < 15; i++) { Wv[i] = (const float*)d_in[1 + 2 * i];  Bv[i] = (const float*)d_in[2 + 2 * i]; }
  for (int i = 0; i < 8;  i++) { We[i] = (const float*)d_in[31 + 2 * i]; Be[i] = (const float*)d_in[32 + 2 * i]; }
  for (int i = 0; i < 6;  i++) { Wc[i] = (const float*)d_in[47 + 2 * i]; Bc[i] = (const float*)d_in[48 + 2 * i]; }
  for (int i = 0; i < 6;  i++) { Wr[i] = (const float*)d_in[59 + 2 * i]; Br[i] = (const float*)d_in[60 + 2 * i]; }
  const float* l2s = (const float*)d_in[71];

  // Workspace carve
  char* wp = (char*)d_ws;
  auto carve = [&](size_t bytes) { char* p = wp; wp += (bytes + 255) & ~(size_t)255; return p; };
  float*  BUF0  = (float*)carve((size_t)B * 64 * 300 * 300 * 4);
  float*  BUF1  = (float*)carve((size_t)B * 64 * 300 * 300 * 4);
  float*  F0raw = (float*)carve((size_t)B * 512 * 38 * 38 * 4);
  float*  F0n   = (float*)carve((size_t)B * 512 * 38 * 38 * 4);
  float*  F1    = (float*)carve((size_t)B * 1024 * 19 * 19 * 4);
  float*  F2    = (float*)carve((size_t)B * 512 * 10 * 10 * 4);
  float*  F3    = (float*)carve((size_t)B * 256 * 5 * 5 * 4);
  float*  F4    = (float*)carve((size_t)B * 256 * 3 * 3 * 4);
  float*  F5    = (float*)carve((size_t)B * 256 * 1 * 1 * 4);
  float*  HSCR  = (float*)carve((size_t)B * 324 * 38 * 38 * 4);
  float*  CLS   = (float*)carve((size_t)B * 8732 * 81 * 4);
  float*  REG   = (float*)carve((size_t)B * 8732 * 4 * 4);
  float*  SCR   = (float*)carve((size_t)B * 8732 * 80 * 4);
  float*  PRI   = (float*)carve((size_t)8732 * 4 * 4);
  float*  CAND  = (float*)carve((size_t)B * CAND_CAP * 6 * 4);
  int*    CNT   = (int*)carve((size_t)B * 4);
  __bf16* WBF   = (__bf16*)carve((size_t)4608 * 1024 * 2);

  auto run_conv = [&](const float* in, float* out, const float* w, const float* b,
                      int Cin, int Hin, int Cout, int Hout,
                      int k, int s, int p, int d, int relu) {
    int K = Cin * k * k;
    int nkw = K * Cout;
    pack_weights<<<(nkw + 255) / 256, 256, 0, stream>>>(w, WBF, K, Cout);
    int M = Hout * Hout;
    int blocks = B * ((M + 127) / 128) * ((Cout + 63) / 64);
    if (k == 1)
      conv_wmma<1><<<blocks, 256, 0, stream>>>(in, WBF, b, out, B, Cin, Hin, Hin,
                                               Cout, Hout, Hout, s, p, d, relu);
    else
      conv_wmma<3><<<blocks, 256, 0, stream>>>(in, WBF, b, out, B, Cin, Hin, Hin,
                                               Cout, Hout, Hout, s, p, d, relu);
  };
  auto run_pool = [&](const float* in, float* out, int C, int Hin, int Hout,
                      int k, int s, int plo) {
    int tot = B * C * Hout * Hout;
    maxpool_k<<<(tot + 255) / 256, 256, 0, stream>>>(in, out, B, C, Hin, Hin,
                                                     Hout, Hout, k, s, plo);
  };

  // ---- VGG trunk ----
  run_conv(x,    BUF0, Wv[0], Bv[0],   3, 300,  64, 300, 3, 1, 1, 1, 1);
  run_conv(BUF0, BUF1, Wv[1], Bv[1],  64, 300,  64, 300, 3, 1, 1, 1, 1);
  run_pool(BUF1, BUF0, 64, 300, 150, 2, 2, 0);
  run_conv(BUF0, BUF1, Wv[2], Bv[2],  64, 150, 128, 150, 3, 1, 1, 1, 1);
  run_conv(BUF1, BUF0, Wv[3], Bv[3], 128, 150, 128, 150, 3, 1, 1, 1, 1);
  run_pool(BUF0, BUF1, 128, 150, 75, 2, 2, 0);
  run_conv(BUF1, BUF0, Wv[4], Bv[4], 128,  75, 256,  75, 3, 1, 1, 1, 1);
  run_conv(BUF0, BUF1, Wv[5], Bv[5], 256,  75, 256,  75, 3, 1, 1, 1, 1);
  run_conv(BUF1, BUF0, Wv[6], Bv[6], 256,  75, 256,  75, 3, 1, 1, 1, 1);
  run_pool(BUF0, BUF1, 256, 75, 38, 2, 2, 0);       // ceil-mode: pad hi=1
  run_conv(BUF1, BUF0, Wv[7], Bv[7], 256,  38, 512,  38, 3, 1, 1, 1, 1);
  run_conv(BUF0, BUF1, Wv[8], Bv[8], 512,  38, 512,  38, 3, 1, 1, 1, 1);
  run_conv(BUF1, F0raw, Wv[9], Bv[9], 512, 38, 512, 38, 3, 1, 1, 1, 1);
  run_pool(F0raw, BUF0, 512, 38, 19, 2, 2, 0);
  run_conv(BUF0, BUF1, Wv[10], Bv[10], 512, 19, 512, 19, 3, 1, 1, 1, 1);
  run_conv(BUF1, BUF0, Wv[11], Bv[11], 512, 19, 512, 19, 3, 1, 1, 1, 1);
  run_conv(BUF0, BUF1, Wv[12], Bv[12], 512, 19, 512, 19, 3, 1, 1, 1, 1);
  run_pool(BUF1, BUF0, 512, 19, 19, 3, 1, 1);
  run_conv(BUF0, BUF1, Wv[13], Bv[13], 512, 19, 1024, 19, 3, 1, 6, 6, 1); // fc6 dil=6
  run_conv(BUF1, F1,   Wv[14], Bv[14], 1024, 19, 1024, 19, 1, 1, 0, 1, 1); // fc7 1x1

  { // L2 norm of conv4_3
    int tot = B * 38 * 38;
    l2norm_k<<<(tot + 255) / 256, 256, 0, stream>>>(F0raw, F0n, l2s, B, 512, 38 * 38);
  }

  // ---- extras ----
  run_conv(F1,   BUF0, We[0], Be[0], 1024, 19, 256, 19, 1, 1, 0, 1, 1);
  run_conv(BUF0, F2,   We[1], Be[1],  256, 19, 512, 10, 3, 2, 1, 1, 1);
  run_conv(F2,   BUF0, We[2], Be[2],  512, 10, 128, 10, 1, 1, 0, 1, 1);
  run_conv(BUF0, F3,   We[3], Be[3],  128, 10, 256,  5, 3, 2, 1, 1, 1);
  run_conv(F3,   BUF0, We[4], Be[4],  256,  5, 128,  5, 1, 1, 0, 1, 1);
  run_conv(BUF0, F4,   We[5], Be[5],  128,  5, 256,  3, 3, 1, 0, 1, 1);
  run_conv(F4,   BUF0, We[6], Be[6],  256,  3, 128,  3, 1, 1, 0, 1, 1);
  run_conv(BUF0, F5,   We[7], Be[7],  128,  3, 256,  1, 3, 1, 0, 1, 1);

  // ---- heads ----
  const float* feats[6] = {F0n, F1, F2, F3, F4, F5};
  const int fC[6]  = {512, 1024, 512, 256, 256, 256};
  const int fH[6]  = {38, 19, 10, 5, 3, 1};
  const int A_[6]  = {4, 6, 6, 6, 4, 4};
  const int OFF[6] = {0, 5776, 7942, 8542, 8692, 8728};
  for (int i = 0; i < 6; i++) {
    // cls head
    run_conv(feats[i], HSCR, Wc[i], Bc[i], fC[i], fH[i], A_[i] * 81, fH[i], 3, 1, 1, 1, 0);
    {
      int tot = B * A_[i] * 81 * fH[i] * fH[i];
      scatter_head<<<(tot + 255) / 256, 256, 0, stream>>>(HSCR, CLS, B, A_[i], 81,
                                                          fH[i], fH[i], OFF[i], 8732);
    }
    // reg head
    run_conv(feats[i], HSCR, Wr[i], Br[i], fC[i], fH[i], A_[i] * 4, fH[i], 3, 1, 1, 1, 0);
    {
      int tot = B * A_[i] * 4 * fH[i] * fH[i];
      scatter_head<<<(tot + 255) / 256, 256, 0, stream>>>(HSCR, REG, B, A_[i], 4,
                                                          fH[i], fH[i], OFF[i], 8732);
    }
  }

  // ---- post-processing ----
  { int tot = B * 8732;
    softmax_k<<<(tot + 255) / 256, 256, 0, stream>>>(CLS, SCR, tot); }
  priors_k<<<(8732 + 255) / 256, 256, 0, stream>>>(PRI);
  hipMemsetAsync(CNT, 0, B * sizeof(int), stream);
  hipMemsetAsync(d_out, 0, (size_t)out_size * sizeof(float), stream);
  { long tot = (long)B * 8732 * 80;
    decode_k<<<(int)((tot + 255) / 256), 256, 0, stream>>>(SCR, REG, PRI, CAND, CNT, B, 8732); }
  float* oboxes   = (float*)d_out;
  float* oscores  = oboxes + (size_t)B * 200 * 4;
  float* oclasses = oscores + (size_t)B * 200;
  nms_k<<<B, 256, 0, stream>>>(CAND, CNT, oboxes, oscores, oclasses);
}
